// UIntxWeightOnlyQuantizedLinear_85152021610531
// MI455X (gfx1250) — matrix operand order repaired and checked by reading, not compiled, not run on hardware
//
#include <hip/hip_runtime.h>

// ---------------------------------------------------------------------------
// out[8192,4096] = x[8192,4096] @ dequant(q[4096,4096])^T + bias
//   dequant: w[n,k] = q * scales[n,k/128] + zeros[n,k/128]
//
// Fast path (needs 96MB workspace):
//   1) x -> f16            (memory-bound pass)
//   2) dequant(q) -> f16   (memory-bound pass)
//   3) f16 WMMA GEMM, tiles DMA'd into LDS by the Tensor Data Mover
//      (tensor_load_to_lds + s_wait_tensorcnt), double-buffered.
// Fallback: fused dequant-in-loop WMMA GEMM (round-1 kernel).
// ---------------------------------------------------------------------------

typedef __attribute__((ext_vector_type(16))) _Float16 v16h;
typedef __attribute__((ext_vector_type(8)))  _Float16 v8h;
typedef __attribute__((ext_vector_type(4)))  _Float16 v4h;
typedef __attribute__((ext_vector_type(8)))  float    v8f;
typedef __attribute__((ext_vector_type(4)))  float    v4f;
typedef __attribute__((ext_vector_type(4)))  int      iv4;
typedef __attribute__((ext_vector_type(8)))  int      iv8;
typedef __attribute__((ext_vector_type(4)))  unsigned int uv4;

constexpr int TOKENS = 8192;
constexpr int NOUT   = 4096;
constexpr int KDIM   = 4096;
constexpr int NGRP   = 32;            // 4096 / 128 quant groups

constexpr int BM = 128, BN = 128;

__device__ __forceinline__ v16h cat8(v8h lo, v8h hi) {
    return __builtin_shufflevector(lo, hi, 0,1,2,3,4,5,6,7,8,9,10,11,12,13,14,15);
}

// ============================ fast path ====================================

constexpr int BK2   = 64;             // K per LDS stage (2 WMMA K-steps)
constexpr int LDSP2 = BK2 + 8;        // padded row stride: 72 halves = 144 B
constexpr int TILEH = BM * LDSP2;     // halves per LDS tile

// Issue one TDM 2-D tile load: rows x cols f16 elements, row stride in
// elements, into LDS at ldsByteOff with 16-B padding after each 128-B row.
__device__ __forceinline__ void tdm_load_tile_f16(const _Float16* gsrc,
                                                  unsigned ldsByteOff,
                                                  int rows, int cols,
                                                  int rowStrideElems)
{
    unsigned long long ga = (unsigned long long)gsrc;
    uv4 g0;
    g0[0] = 1u;                                     // count=1, user descriptor
    g0[1] = ldsByteOff;                             // LDS destination (bytes)
    g0[2] = (unsigned)(ga & 0xFFFFFFFFu);           // global_addr[31:0]
    g0[3] = (unsigned)((ga >> 32) & 0x01FFFFFFu)    // global_addr[56:32]
          | (2u << 30);                             // type = 2 ("image")

    const unsigned td0 = 1u << 20, td1 = 1u << 20;  // tensor dims: large (in-bounds)
    iv8 g1;
    g1[0] = (int)((1u << 16)        // data_size = 2 bytes
                | (1u << 20)        // pad_enable
                | (4u << 22)        // pad_interval: every 32 DWORDs (=128B row)
                | (3u << 25));      // pad_amount: 4 DWORDs (=16B)
    g1[1] = (int)((td0 & 0xFFFFu) << 16);                    // tensor_dim0 lo
    g1[2] = (int)((td0 >> 16) | ((td1 & 0xFFFFu) << 16));    // dim0 hi | dim1 lo
    g1[3] = (int)((td1 >> 16) | ((unsigned)cols << 16));     // dim1 hi | tile_dim0
    g1[4] = rows;                                            // tile_dim1 (dim2=0)
    g1[5] = rowStrideElems;                                  // tensor_dim0_stride
    g1[6] = 0;
    g1[7] = 0;

    iv4 z4 = {0, 0, 0, 0};
#if defined(__clang_major__) && (__clang_major__ >= 23)
    iv8 z8 = {0, 0, 0, 0, 0, 0, 0, 0};
    __builtin_amdgcn_tensor_load_to_lds(g0, g1, z4, z4, z8, 0);
#else
    __builtin_amdgcn_tensor_load_to_lds(g0, g1, z4, z4, 0);
#endif
}

__global__ __launch_bounds__(256)
void qlinear_xf32_to_f16(const float* __restrict__ x, _Float16* __restrict__ xh)
{
    const size_t i = ((size_t)blockIdx.x * blockDim.x + threadIdx.x) * 8;
    v4f a = *(const v4f*)&x[i];
    v4f b = *(const v4f*)&x[i + 4];
    v8h h;
    #pragma unroll
    for (int j = 0; j < 4; ++j) { h[j] = (_Float16)a[j]; h[4 + j] = (_Float16)b[j]; }
    *(v8h*)&xh[i] = h;
}

__global__ __launch_bounds__(256)
void qlinear_dequant_w_f16(const int* __restrict__ qw,
                           const float* __restrict__ scales,
                           const float* __restrict__ zeros,
                           _Float16* __restrict__ wh)
{
    const size_t i = ((size_t)blockIdx.x * blockDim.x + threadIdx.x) * 8;
    const int row = (int)(i / KDIM);
    const int col = (int)(i % KDIM);
    const int g   = col >> 7;                       // 8 cols stay in one group
    const float s = scales[row * NGRP + g];
    const float z = zeros [row * NGRP + g];
    iv4 a = *(const iv4*)&qw[i];
    iv4 b = *(const iv4*)&qw[i + 4];
    v8h h;
    #pragma unroll
    for (int j = 0; j < 4; ++j) {
        h[j]     = (_Float16)((float)a[j] * s + z);
        h[4 + j] = (_Float16)((float)b[j] * s + z);
    }
    *(v8h*)&wh[i] = h;
}

__global__ __launch_bounds__(256)
void qlinear_f16_gemm_tdm(const _Float16* __restrict__ xh,
                          const _Float16* __restrict__ wh,
                          const float* __restrict__ bias,
                          float* __restrict__ out)
{
    // [buf0: A, B][buf1: A, B] -> 72 KB total, 4 workgroups per WGP
    __shared__ __align__(16) _Float16 lds[4 * TILEH];

    const int tid  = threadIdx.x;
    const int lane = tid & 31;
    const int wave = tid >> 5;
    const int hl   = lane >> 4;
    const int lrow = lane & 15;

    const int tBase = blockIdx.y * BM;
    const int nBase = blockIdx.x * BN;
    const int wm = (wave & 3) * 32;
    const int wn = (wave >> 2) * 64;

    const unsigned ldsBase = (unsigned)(unsigned long long)(&lds[0]);

    v8f acc[2][4];
    #pragma unroll
    for (int mt = 0; mt < 2; ++mt)
        #pragma unroll
        for (int nt = 0; nt < 4; ++nt)
            acc[mt][nt] = v8f{};

    const _Float16* xg = xh + (size_t)tBase * KDIM;
    const _Float16* wg = wh + (size_t)nBase * KDIM;

    // prologue: DMA first K-tile pair into buffer 0
    if (wave == 0) {
        tdm_load_tile_f16(xg, ldsBase + 0u * TILEH * 2u, BM, BK2, KDIM);
        tdm_load_tile_f16(wg, ldsBase + 1u * TILEH * 2u, BN, BK2, KDIM);
    }

    constexpr int NIT = KDIM / BK2;   // 64
    for (int it = 0; it < NIT; ++it) {
        const int buf = it & 1;
        if (wave == 0) __builtin_amdgcn_s_wait_tensorcnt(0);
        __syncthreads();              // LDS[buf] ready for everyone

        // overlap: DMA next tile pair into the other buffer while we compute
        if (wave == 0 && it + 1 < NIT) {
            const int kk = (it + 1) * BK2;
            const unsigned nb = (unsigned)(buf ^ 1);
            tdm_load_tile_f16(xg + kk, ldsBase + (2u*nb + 0u) * TILEH * 2u, BM, BK2, KDIM);
            tdm_load_tile_f16(wg + kk, ldsBase + (2u*nb + 1u) * TILEH * 2u, BN, BK2, KDIM);
        }

        const _Float16* lA = &lds[(2 * buf + 0) * TILEH];
        const _Float16* lB = &lds[(2 * buf + 1) * TILEH];

        #pragma unroll
        for (int ks = 0; ks < 2; ++ks) {
            // wave32 A layout: lanes 0-15 K{0..7,16..23}, lanes 16-31 K{8..15,24..31}
            v16h af[2];
            #pragma unroll
            for (int mt = 0; mt < 2; ++mt) {
                const int base = (wm + mt * 16 + lrow) * LDSP2 + ks * 32;
                v8h lo = *(const v8h*)&lA[base + 8 * hl];
                v8h hi = *(const v8h*)&lA[base + 16 + 8 * hl];
                af[mt] = cat8(lo, hi);
            }
            // B layout: lane = N column, lanes 0-15 K{0..7,16..23}, lanes 16-31 K{8..15,24..31}
            v16h bf[4];
            #pragma unroll
            for (int nt = 0; nt < 4; ++nt) {
                const int base = (wn + nt * 16 + lrow) * LDSP2 + ks * 32;
                v8h lo = *(const v8h*)&lB[base + 8 * hl];
                v8h hi = *(const v8h*)&lB[base + 16 + 8 * hl];
                bf[nt] = cat8(lo, hi);
            }
            #pragma unroll
            for (int mt = 0; mt < 2; ++mt)
                #pragma unroll
                for (int nt = 0; nt < 4; ++nt)
                    acc[mt][nt] = __builtin_amdgcn_wmma_f32_16x16x32_f16(
                        false, af[mt], false, bf[nt],
                        (short)0, acc[mt][nt], false, false);
        }
        __syncthreads();
    }

    float bv[4];
    #pragma unroll
    for (int nt = 0; nt < 4; ++nt)
        bv[nt] = bias[nBase + wn + nt * 16 + lrow];

    #pragma unroll
    for (int mt = 0; mt < 2; ++mt)
        #pragma unroll
        for (int nt = 0; nt < 4; ++nt) {
            const int n = nBase + wn + nt * 16 + lrow;
            #pragma unroll
            for (int r = 0; r < 8; ++r) {
                const int t = tBase + wm + mt * 16 + r + 8 * hl;
                out[(size_t)t * NOUT + n] = acc[mt][nt][r] + bv[nt];
            }
        }
}

// ===================== fallback: fused dequant GEMM ========================

constexpr int BKF   = 32;
constexpr int LDSPF = BKF + 8;

__global__ __launch_bounds__(256)
void qlinear_wmma_fused(const float* __restrict__ x,
                        const int*   __restrict__ qw,
                        const float* __restrict__ scales,
                        const float* __restrict__ zeros,
                        const float* __restrict__ bias,
                        float*       __restrict__ out)
{
    __shared__ __align__(16) _Float16 lsA[BM * LDSPF];
    __shared__ __align__(16) _Float16 lsB[BN * LDSPF];

    const int tid  = threadIdx.x;
    const int lane = tid & 31;
    const int wave = tid >> 5;
    const int hl   = lane >> 4;
    const int lrow = lane & 15;

    const int tBase = blockIdx.y * BM;
    const int nBase = blockIdx.x * BN;
    const int wm = (wave & 3) * 32;
    const int wn = (wave >> 2) * 64;

    const int srow = tid >> 3;
    const int scol = (tid & 7) * 4;

    v8f acc[2][4];
    #pragma unroll
    for (int mt = 0; mt < 2; ++mt)
        #pragma unroll
        for (int nt = 0; nt < 4; ++nt)
            acc[mt][nt] = v8f{};

    for (int kk = 0; kk < KDIM; kk += BKF) {
        const int g = kk >> 7;

        #pragma unroll
        for (int i = 0; i < 4; ++i) {
            const int r = srow + i * 32;
            v4f xa = *(const v4f*)&x[(size_t)(tBase + r) * KDIM + kk + scol];
            v4h ha;
            #pragma unroll
            for (int j = 0; j < 4; ++j) ha[j] = (_Float16)xa[j];
            *(v4h*)&lsA[r * LDSPF + scol] = ha;
        }
        #pragma unroll
        for (int i = 0; i < 4; ++i) {
            const int r = srow + i * 32;
            iv4 qa = *(const iv4*)&qw[(size_t)(nBase + r) * KDIM + kk + scol];
            const float s = scales[(nBase + r) * NGRP + g];
            const float z = zeros [(nBase + r) * NGRP + g];
            v4h hb;
            #pragma unroll
            for (int j = 0; j < 4; ++j) hb[j] = (_Float16)((float)qa[j] * s + z);
            *(v4h*)&lsB[r * LDSPF + scol] = hb;
        }
        __syncthreads();

        v16h afrag[2];
        #pragma unroll
        for (int mt = 0; mt < 2; ++mt) {
            const int r = wm + mt * 16 + lrow;
            v8h lo = *(const v8h*)&lsA[r * LDSPF + 8 * hl];
            v8h hi = *(const v8h*)&lsA[r * LDSPF + 16 + 8 * hl];
            afrag[mt] = cat8(lo, hi);
        }
        v16h bfrag[4];
        #pragma unroll
        for (int nt = 0; nt < 4; ++nt) {
            const int r = wn + nt * 16 + lrow;
            v8h lo = *(const v8h*)&lsB[r * LDSPF + 8 * hl];
            v8h hi = *(const v8h*)&lsB[r * LDSPF + 16 + 8 * hl];
            bfrag[nt] = cat8(lo, hi);
        }
        #pragma unroll
        for (int mt = 0; mt < 2; ++mt)
            #pragma unroll
            for (int nt = 0; nt < 4; ++nt)
                acc[mt][nt] = __builtin_amdgcn_wmma_f32_16x16x32_f16(
                    false, afrag[mt], false, bfrag[nt],
                    (short)0, acc[mt][nt], false, false);
        __syncthreads();
    }

    float bv[4];
    #pragma unroll
    for (int nt = 0; nt < 4; ++nt)
        bv[nt] = bias[nBase + wn + nt * 16 + lrow];

    #pragma unroll
    for (int mt = 0; mt < 2; ++mt)
        #pragma unroll
        for (int nt = 0; nt < 4; ++nt) {
            const int n = nBase + wn + nt * 16 + lrow;
            #pragma unroll
            for (int r = 0; r < 8; ++r) {
                const int t = tBase + wm + mt * 16 + r + 8 * hl;
                out[(size_t)t * NOUT + n] = acc[mt][nt][r] + bv[nt];
            }
        }
}

// ============================== launcher ===================================

extern "C" void kernel_launch(void* const* d_in, const int* in_sizes, int n_in,
                              void* d_out, int out_size, void* d_ws, size_t ws_size,
                              hipStream_t stream) {
    (void)in_sizes; (void)n_in; (void)out_size;
    const float* x      = (const float*)d_in[0];
    const int*   qw     = (const int*)  d_in[1];
    const float* scales = (const float*)d_in[2];
    const float* zeros  = (const float*)d_in[3];
    const float* bias   = (const float*)d_in[4];
    float* out = (float*)d_out;

    const size_t xBytes = (size_t)TOKENS * KDIM * sizeof(_Float16);  // 64 MB
    const size_t wBytes = (size_t)NOUT   * KDIM * sizeof(_Float16);  // 32 MB

    dim3 grid(NOUT / BN, TOKENS / BM);   // (32, 64)
    dim3 block(256);

    if (ws_size >= xBytes + wBytes) {
        _Float16* xh = (_Float16*)d_ws;
        _Float16* wh = (_Float16*)((char*)d_ws + xBytes);
        qlinear_xf32_to_f16<<<(TOKENS * KDIM / 8) / 256, 256, 0, stream>>>(x, xh);
        qlinear_dequant_w_f16<<<(NOUT * KDIM / 8) / 256, 256, 0, stream>>>(
            qw, scales, zeros, wh);
        qlinear_f16_gemm_tdm<<<grid, block, 0, stream>>>(xh, wh, bias, out);
    } else {
        qlinear_wmma_fused<<<grid, block, 0, stream>>>(
            x, qw, scales, zeros, bias, out);
    }
}
